// ViTMoEBlock_944892805333
// MI455X (gfx1250) — matrix operand order repaired and checked by reading, not compiled, not run on hardware
//
#include <hip/hip_runtime.h>
#include <hip/hip_bf16.h>

typedef __bf16 bf16;
typedef __bf16 bf16x16 __attribute__((ext_vector_type(16)));
typedef __bf16 bf16x8  __attribute__((ext_vector_type(8)));
typedef float  f32x8   __attribute__((ext_vector_type(8)));
typedef uint32_t u32x4 __attribute__((ext_vector_type(4)));
typedef uint32_t u32x8 __attribute__((ext_vector_type(8)));

#define NB   32      // batch
#define HH   14
#define WW   14
#define DD   768
#define NH   12
#define HD   64
#define FF   3072
#define NE   8
#define NT   (NB*HH*WW)   // 6272 tokens
#define TPS  (HH*WW)      // 196 tokens per sample
#define MT_PER_S 13       // ceil(196/16)

// ---------------- workspace layout (bytes) ----------------
static constexpr size_t SZ_W    = (size_t)DD*DD*2;        // one 768x768 bf16 weight
static constexpr size_t SZ_WE   = (size_t)NE*DD*FF*2;     // all experts bf16
static constexpr size_t SZ_ACT  = (size_t)NT*DD*2;        // token x D bf16
static constexpr size_t OFF_WQ  = 0;
static constexpr size_t OFF_WK  = OFF_WQ + SZ_W;
static constexpr size_t OFF_WV  = OFF_WK + SZ_W;
static constexpr size_t OFF_WO  = OFF_WV + SZ_W;
static constexpr size_t OFF_WE1 = OFF_WO + SZ_W;
static constexpr size_t OFF_WE2 = OFF_WE1 + SZ_WE;
static constexpr size_t OFF_NX  = OFF_WE2 + SZ_WE;
static constexpr size_t OFF_Q   = OFF_NX + SZ_ACT;
static constexpr size_t OFF_K   = OFF_Q + SZ_ACT;
static constexpr size_t OFF_V   = OFF_K + SZ_ACT;
static constexpr size_t OFF_O   = OFF_V + SZ_ACT;
static constexpr size_t OFF_X1  = OFF_O + SZ_ACT;                 // f32
static constexpr size_t OFF_NX2 = OFF_X1 + (size_t)NT*DD*4;
static constexpr size_t OFF_H   = OFF_NX2 + SZ_ACT;               // bf16 NT x F
static constexpr size_t OFF_POOL= OFF_H + (size_t)NT*FF*2;        // f32 B x D
static constexpr size_t OFF_WBE = OFF_POOL + (size_t)NB*DD*4;     // f32 B x E

// ---------------- small helpers ----------------
__device__ __forceinline__ float gelu_exact(float x) {
    return 0.5f * x * (1.0f + erff(x * 0.70710678118654752440f));
}

// Tensor Data Mover: load a 16-row x kchunk-col bf16 tile (row stride =
// stride_elems) from global into LDS at lds_off. 2D descriptor, 2-group form
// (VADDR2/3 = NULL). Tracked by TENSORcnt. Issue from ONE wave only.
__device__ __forceinline__ void tdm_load_tile16(const bf16* gsrc, uint32_t lds_off,
                                                uint32_t stride_elems, uint32_t kchunk) {
    uint64_t ga = (uint64_t)(uintptr_t)gsrc;
    u32x4 g0;
    g0[0] = 1u;                                           // count=1, user descriptor
    g0[1] = lds_off;                                      // lds_addr (bytes)
    g0[2] = (uint32_t)ga;                                 // global_addr[31:0]
    g0[3] = (uint32_t)((ga >> 32) & 0x01FFFFFFu)          // global_addr[56:32]
            | (2u << 30);                                 // type = 2 ("image")
    u32x8 g1;
    g1[0] = (1u << 16);                                   // wg_mask=0, data_size=1 (2B)
    g1[1] = (kchunk & 0xFFFFu) << 16;                     // tensor_dim0[15:0]
    g1[2] = (kchunk >> 16) | (16u << 16);                 // tensor_dim0[31:16] | tensor_dim1[15:0]=16
    g1[3] = (kchunk & 0xFFFFu) << 16;                     // tensor_dim1[31:16]=0 | tile_dim0=kchunk
    g1[4] = 16u;                                          // tile_dim1=16, tile_dim2=0
    g1[5] = stride_elems;                                 // tensor_dim0_stride[31:0]
    g1[6] = 0u;                                           // stride[47:32]=0, dim1_stride lo=0
    g1[7] = 0u;
    asm volatile("tensor_load_to_lds %0, %1" :: "s"(g0), "s"(g1) : "memory");
}

// wave-level 16(M) x 64(N) WMMA strip over one K-chunk; A comes from the LDS
// tile (rows contiguous, stride = kchunk), B from global (row-major, ldb).
__device__ __forceinline__ void gemm_tile_compute(
    const bf16* __restrict__ sA, int kchunk,
    const bf16* __restrict__ B, int ldb,
    int n0, int kbase, f32x8 acc[4], int lane)
{
    const int r  = lane & 15;
    const int kb = (lane >> 4) * 8;   // K sub-half select per ISA A layout
    for (int k0 = 0; k0 < kchunk; k0 += 32) {
        const bf16* ap = sA + r * kchunk + k0 + kb;
        bf16x8 alo = *(const bf16x8*)(ap);           // ds_load_b128
        bf16x8 ahi = *(const bf16x8*)(ap + 16);      // ds_load_b128
        bf16x16 a;
#pragma unroll
        for (int j = 0; j < 8; ++j) { a[j] = alo[j]; a[8 + j] = ahi[j]; }
        const bf16* bbase = B + (size_t)(kbase + k0 + kb) * ldb + n0 + r;
        __builtin_prefetch(bbase + (size_t)32 * ldb, 0, 0);   // next B K-chunk -> GL2
#pragma unroll
        for (int s = 0; s < 4; ++s) {
            bf16x16 bfrag;
#pragma unroll
            for (int j = 0; j < 16; ++j) bfrag[j] = bbase[(size_t)(j < 8 ? j : j + 8) * ldb + s * 16];
            acc[s] = __builtin_amdgcn_wmma_f32_16x16x32_bf16(
                false, a, false, bfrag, (short)0, acc[s], false, false);
        }
    }
}

// ---------------- kernels ----------------
__global__ void cvt_bf16_kernel(const float* __restrict__ s, bf16* __restrict__ d, int n) {
    int i = blockIdx.x * blockDim.x + threadIdx.x;
    int stride = gridDim.x * blockDim.x;
    for (; i < n; i += stride) d[i] = (bf16)s[i];
}

__global__ void copy_f32_kernel(const float* __restrict__ s, float* __restrict__ d, int n) {
    int i = blockIdx.x * blockDim.x + threadIdx.x;
    int stride = gridDim.x * blockDim.x;
    for (; i < n; i += stride) d[i] = s[i];
}

// layer norm over D=768, one 256-thread block per token, bf16 output
__global__ void ln_bf16_kernel(const float* __restrict__ x, const float* __restrict__ g,
                               const float* __restrict__ b, bf16* __restrict__ out) {
    const int t = blockIdx.x;
    const int tid = threadIdx.x;
    const float* xr = x + (size_t)t * DD;
    __shared__ float red[256];
    float v0 = xr[tid], v1 = xr[tid + 256], v2 = xr[tid + 512];
    red[tid] = v0 + v1 + v2;
    __syncthreads();
    for (int s = 128; s > 0; s >>= 1) { if (tid < s) red[tid] += red[tid + s]; __syncthreads(); }
    float mean = red[0] * (1.0f / DD);
    __syncthreads();
    float d0 = v0 - mean, d1 = v1 - mean, d2 = v2 - mean;
    red[tid] = d0 * d0 + d1 * d1 + d2 * d2;
    __syncthreads();
    for (int s = 128; s > 0; s >>= 1) { if (tid < s) red[tid] += red[tid + s]; __syncthreads(); }
    float rstd = rsqrtf(red[0] * (1.0f / DD) + 1e-5f);
    bf16* orow = out + (size_t)t * DD;
    orow[tid]       = (bf16)(d0 * rstd * g[tid]       + b[tid]);
    orow[tid + 256] = (bf16)(d1 * rstd * g[tid + 256] + b[tid + 256]);
    orow[tid + 512] = (bf16)(d2 * rstd * g[tid + 512] + b[tid + 512]);
}

// out_bf16[M x 768] = A_bf16[M x 768] @ W_bf16[768 x 768] + bias   (Q/K/V proj)
__global__ void gemm_proj_kernel(const bf16* __restrict__ A, const bf16* __restrict__ Wt,
                                 const float* __restrict__ bias, bf16* __restrict__ out) {
    __shared__ bf16 sA[16 * DD];
    const int lane = threadIdx.x & 31, wave = threadIdx.x >> 5;
    const int m0 = blockIdx.x * 16;
    const int n0 = blockIdx.y * 256 + wave * 64;
    if (wave == 0) {
        tdm_load_tile16(A + (size_t)m0 * DD, (uint32_t)(uintptr_t)&sA[0], DD, DD);
        __builtin_amdgcn_s_wait_tensorcnt(0);
    }
    __syncthreads();
    f32x8 acc[4];
    const f32x8 fz = {0.f,0.f,0.f,0.f,0.f,0.f,0.f,0.f};
    acc[0]=fz; acc[1]=fz; acc[2]=fz; acc[3]=fz;
    gemm_tile_compute(sA, DD, Wt, DD, n0, 0, acc, lane);
    const int col = lane & 15, rbase = (lane >> 4) * 8;
#pragma unroll
    for (int s = 0; s < 4; ++s) {
        int n = n0 + s * 16 + col;
        float bn = bias[n];
#pragma unroll
        for (int v = 0; v < 8; ++v) {
            int m = m0 + rbase + v;
            out[(size_t)m * DD + n] = (bf16)(acc[s][v] + bn);
        }
    }
}

// x1_f32 = x_res + o_bf16 @ Wo_bf16 + bo
__global__ void gemm_attnout_kernel(const bf16* __restrict__ A, const bf16* __restrict__ Wt,
                                    const float* __restrict__ bias, const float* __restrict__ xres,
                                    float* __restrict__ x1) {
    __shared__ bf16 sA[16 * DD];
    const int lane = threadIdx.x & 31, wave = threadIdx.x >> 5;
    const int m0 = blockIdx.x * 16;
    const int n0 = blockIdx.y * 256 + wave * 64;
    if (wave == 0) {
        tdm_load_tile16(A + (size_t)m0 * DD, (uint32_t)(uintptr_t)&sA[0], DD, DD);
        __builtin_amdgcn_s_wait_tensorcnt(0);
    }
    __syncthreads();
    f32x8 acc[4];
    const f32x8 fz = {0.f,0.f,0.f,0.f,0.f,0.f,0.f,0.f};
    acc[0]=fz; acc[1]=fz; acc[2]=fz; acc[3]=fz;
    gemm_tile_compute(sA, DD, Wt, DD, n0, 0, acc, lane);
    const int col = lane & 15, rbase = (lane >> 4) * 8;
#pragma unroll
    for (int s = 0; s < 4; ++s) {
        int n = n0 + s * 16 + col;
        float bn = bias[n];
#pragma unroll
        for (int v = 0; v < 8; ++v) {
            size_t idx = (size_t)(m0 + rbase + v) * DD + n;
            x1[idx] = xres[idx] + acc[s][v] + bn;
        }
    }
}

// row-wise attention: one block per (row, head, sample), seq len 14, head dim 64
__global__ void attn_kernel(const bf16* __restrict__ qb, const bf16* __restrict__ kb,
                            const bf16* __restrict__ vb, bf16* __restrict__ ob) {
    const int hrow = blockIdx.x, head = blockIdx.y, b = blockIdx.z;
    const int d = threadIdx.x;                     // 0..63
    const int tokBase = (b * HH + hrow) * WW;
    const int col = head * HD + d;
    __shared__ float qs[WW * HD], ks[WW * HD], vs[WW * HD];
    __shared__ float sco[WW], sex[WW];
    for (int w = 0; w < WW; ++w) {
        size_t idx = (size_t)(tokBase + w) * DD + col;
        qs[w * HD + d] = (float)qb[idx];
        ks[w * HD + d] = (float)kb[idx];
        vs[w * HD + d] = (float)vb[idx];
    }
    __syncthreads();
    for (int i = 0; i < WW; ++i) {
        if (d < WW) {
            float s = 0.f;
            for (int dd = 0; dd < HD; ++dd) s += qs[i * HD + dd] * ks[d * HD + dd];
            sco[d] = s * 0.125f;   // 1/sqrt(64)
        }
        __syncthreads();
        if (d < WW) {
            float mx = sco[0];
            for (int j = 1; j < WW; ++j) mx = fmaxf(mx, sco[j]);
            sex[d] = expf(sco[d] - mx);
        }
        __syncthreads();
        float denom = 0.f;
        for (int j = 0; j < WW; ++j) denom += sex[j];
        float o = 0.f;
        for (int j = 0; j < WW; ++j) o += sex[j] * vs[j * HD + d];
        ob[(size_t)(tokBase + i) * DD + col] = (bf16)(o / denom);
        __syncthreads();
    }
}

// pooled[b][d] = mean over 196 tokens of nx2
__global__ void pool_kernel(const bf16* __restrict__ nx2, float* __restrict__ pooled) {
    int idx = blockIdx.x * blockDim.x + threadIdx.x;   // 32*768 threads
    if (idx >= NB * DD) return;
    int b = idx / DD, dd = idx - b * DD;
    float s = 0.f;
    for (int t = 0; t < TPS; ++t) s += (float)nx2[(size_t)(b * TPS + t) * DD + dd];
    pooled[idx] = s * (1.0f / TPS);
}

// router: 1 block, thread b handles sample b
__global__ void router_kernel(const float* __restrict__ pooled, const float* __restrict__ Wg,
                              const float* __restrict__ bg, float* __restrict__ probs_out,
                              int* __restrict__ topi_out, float* __restrict__ wbe) {
    const int b = threadIdx.x;
    if (b >= NB) return;
    float logit[NE];
#pragma unroll
    for (int e = 0; e < NE; ++e) logit[e] = bg[e];
    for (int dd = 0; dd < DD; ++dd) {
        float p = pooled[b * DD + dd];
#pragma unroll
        for (int e = 0; e < NE; ++e) logit[e] += p * Wg[dd * NE + e];
    }
    float mx = logit[0];
#pragma unroll
    for (int e = 1; e < NE; ++e) mx = fmaxf(mx, logit[e]);
    float pr[NE]; float sum = 0.f;
#pragma unroll
    for (int e = 0; e < NE; ++e) { pr[e] = expf(logit[e] - mx); sum += pr[e]; }
#pragma unroll
    for (int e = 0; e < NE; ++e) { pr[e] /= sum; probs_out[b * NE + e] = pr[e]; }
    int i0 = 0;
#pragma unroll
    for (int e = 1; e < NE; ++e) if (pr[e] > pr[i0]) i0 = e;
    int i1 = (i0 == 0) ? 1 : 0;
#pragma unroll
    for (int e = 0; e < NE; ++e) if (e != i0 && pr[e] > pr[i1]) i1 = e;
    float s2 = pr[i0] + pr[i1] + 1e-8f;
#pragma unroll
    for (int e = 0; e < NE; ++e) wbe[b * NE + e] = 0.f;
    wbe[b * NE + i0] = pr[i0] / s2;
    wbe[b * NE + i1] += pr[i1] / s2;
    topi_out[b * 2 + 0] = i0;
    topi_out[b * 2 + 1] = i1;
}

// expert GEMM1: h = gelu(nx2 @ We1[e] + be1[e]); per-sample M tiling + early exit
__global__ void gemm_moe1_kernel(const bf16* __restrict__ A, const bf16* __restrict__ We1e,
                                 const float* __restrict__ be1e, bf16* __restrict__ h,
                                 const float* __restrict__ wbe, int e) {
    __shared__ bf16 sA[16 * DD];
    const int b = blockIdx.x / MT_PER_S, mt = blockIdx.x % MT_PER_S;
    if (wbe[b * NE + e] == 0.0f) return;           // block-uniform exit (before barrier)
    const int lane = threadIdx.x & 31, wave = threadIdx.x >> 5;
    const int n0 = blockIdx.y * 256 + wave * 64;
    const size_t row0 = (size_t)(b * TPS + mt * 16);   // edge rows read scratch; stores guarded
    if (wave == 0) {
        tdm_load_tile16(A + row0 * DD, (uint32_t)(uintptr_t)&sA[0], DD, DD);
        __builtin_amdgcn_s_wait_tensorcnt(0);
    }
    __syncthreads();
    f32x8 acc[4];
    const f32x8 fz = {0.f,0.f,0.f,0.f,0.f,0.f,0.f,0.f};
    acc[0]=fz; acc[1]=fz; acc[2]=fz; acc[3]=fz;
    gemm_tile_compute(sA, DD, We1e, FF, n0, 0, acc, lane);
    const int col = lane & 15, rbase = (lane >> 4) * 8;
#pragma unroll
    for (int s = 0; s < 4; ++s) {
        int n = n0 + s * 16 + col;
        float bn = be1e[n];
#pragma unroll
        for (int v = 0; v < 8; ++v) {
            int mloc = mt * 16 + rbase + v;
            if (mloc < TPS)
                h[(size_t)(b * TPS + mloc) * FF + n] = (bf16)gelu_exact(acc[s][v] + bn);
        }
    }
}

// expert GEMM2: xout += wbe[b][e] * (h @ We2[e] + be2[e]); K=3072 in 4 TDM chunks
__global__ void gemm_moe2_kernel(const bf16* __restrict__ h, const bf16* __restrict__ We2e,
                                 const float* __restrict__ be2e, float* __restrict__ xout,
                                 const float* __restrict__ wbe, int e) {
    __shared__ bf16 sA[16 * DD];
    const int b = blockIdx.x / MT_PER_S, mt = blockIdx.x % MT_PER_S;
    const float w = wbe[b * NE + e];
    if (w == 0.0f) return;                          // block-uniform exit (before barrier)
    const int lane = threadIdx.x & 31, wave = threadIdx.x >> 5;
    const int n0 = blockIdx.y * 256 + wave * 64;
    const size_t row0 = (size_t)(b * TPS + mt * 16);
    f32x8 acc[4];
    const f32x8 fz = {0.f,0.f,0.f,0.f,0.f,0.f,0.f,0.f};
    acc[0]=fz; acc[1]=fz; acc[2]=fz; acc[3]=fz;
    for (int kc = 0; kc < FF; kc += DD) {
        if (wave == 0) {
            tdm_load_tile16(h + row0 * FF + kc, (uint32_t)(uintptr_t)&sA[0], FF, DD);
            __builtin_amdgcn_s_wait_tensorcnt(0);
        }
        __syncthreads();
        gemm_tile_compute(sA, DD, We2e, DD, n0, kc, acc, lane);
        __syncthreads();                            // protect sA before next TDM overwrite
    }
    const int col = lane & 15, rbase = (lane >> 4) * 8;
#pragma unroll
    for (int s = 0; s < 4; ++s) {
        int n = n0 + s * 16 + col;
        float bn = be2e[n];
#pragma unroll
        for (int v = 0; v < 8; ++v) {
            int mloc = mt * 16 + rbase + v;
            if (mloc < TPS) {
                size_t idx = (size_t)(b * TPS + mloc) * DD + n;
                xout[idx] += w * (acc[s][v] + bn);  // expert launches serialize on stream
            }
        }
    }
}

// ---------------- host ----------------
extern "C" void kernel_launch(void* const* d_in, const int* in_sizes, int n_in,
                              void* d_out, int out_size, void* d_ws, size_t ws_size,
                              hipStream_t stream) {
    const float* x   = (const float*)d_in[0];
    const float* g1  = (const float*)d_in[1];
    const float* b1  = (const float*)d_in[2];
    const float* Wq  = (const float*)d_in[3];
    const float* bq  = (const float*)d_in[4];
    const float* Wk  = (const float*)d_in[5];
    const float* bk  = (const float*)d_in[6];
    const float* Wv  = (const float*)d_in[7];
    const float* bv  = (const float*)d_in[8];
    const float* Wo  = (const float*)d_in[9];
    const float* bo  = (const float*)d_in[10];
    const float* g2  = (const float*)d_in[11];
    const float* b2  = (const float*)d_in[12];
    const float* Wg  = (const float*)d_in[13];
    const float* bg  = (const float*)d_in[14];
    const float* We1 = (const float*)d_in[15];
    const float* be1 = (const float*)d_in[16];
    const float* We2 = (const float*)d_in[17];
    const float* be2 = (const float*)d_in[18];

    char* ws = (char*)d_ws;
    bf16* wq_bf  = (bf16*)(ws + OFF_WQ);
    bf16* wk_bf  = (bf16*)(ws + OFF_WK);
    bf16* wv_bf  = (bf16*)(ws + OFF_WV);
    bf16* wo_bf  = (bf16*)(ws + OFF_WO);
    bf16* we1_bf = (bf16*)(ws + OFF_WE1);
    bf16* we2_bf = (bf16*)(ws + OFF_WE2);
    bf16* nx_bf  = (bf16*)(ws + OFF_NX);
    bf16* q_bf   = (bf16*)(ws + OFF_Q);
    bf16* k_bf   = (bf16*)(ws + OFF_K);
    bf16* v_bf   = (bf16*)(ws + OFF_V);
    bf16* o_bf   = (bf16*)(ws + OFF_O);
    float* x1    = (float*)(ws + OFF_X1);
    bf16* nx2_bf = (bf16*)(ws + OFF_NX2);
    bf16* h_bf   = (bf16*)(ws + OFF_H);
    float* pooled= (float*)(ws + OFF_POOL);
    float* wbe   = (float*)(ws + OFF_WBE);

    const int X_ELEMS = NT * DD;                     // 4,816,896
    float* xout      = (float*)d_out;
    float* probs_out = (float*)d_out + X_ELEMS;
    int*   topi_out  = (int*)d_out + X_ELEMS + NB * NE;

    // 1) weight conversions fp32 -> bf16 (every call; deterministic)
    cvt_bf16_kernel<<<512, 256, 0, stream>>>(Wq, wq_bf, DD * DD);
    cvt_bf16_kernel<<<512, 256, 0, stream>>>(Wk, wk_bf, DD * DD);
    cvt_bf16_kernel<<<512, 256, 0, stream>>>(Wv, wv_bf, DD * DD);
    cvt_bf16_kernel<<<512, 256, 0, stream>>>(Wo, wo_bf, DD * DD);
    cvt_bf16_kernel<<<4096, 256, 0, stream>>>(We1, we1_bf, NE * DD * FF);
    cvt_bf16_kernel<<<4096, 256, 0, stream>>>(We2, we2_bf, NE * FF * DD);

    // 2) LN1 -> bf16
    ln_bf16_kernel<<<NT, 256, 0, stream>>>(x, g1, b1, nx_bf);

    // 3) Q/K/V projections (TDM + WMMA)
    dim3 gProj(NT / 16, DD / 256);                   // (392, 3)
    gemm_proj_kernel<<<gProj, 128, 0, stream>>>(nx_bf, wq_bf, bq, q_bf);
    gemm_proj_kernel<<<gProj, 128, 0, stream>>>(nx_bf, wk_bf, bk, k_bf);
    gemm_proj_kernel<<<gProj, 128, 0, stream>>>(nx_bf, wv_bf, bv, v_bf);

    // 4) row-wise attention
    attn_kernel<<<dim3(HH, NH, NB), 64, 0, stream>>>(q_bf, k_bf, v_bf, o_bf);

    // 5) output projection + residual -> x1 (f32)
    gemm_attnout_kernel<<<gProj, 128, 0, stream>>>(o_bf, wo_bf, bo, x, x1);

    // 6) LN2 -> bf16
    ln_bf16_kernel<<<NT, 256, 0, stream>>>(x1, g2, b2, nx2_bf);

    // 7) global pool + router (writes all_probs / top_i outputs and wbe)
    pool_kernel<<<(NB * DD + 255) / 256, 256, 0, stream>>>(nx2_bf, pooled);
    router_kernel<<<1, 32, 0, stream>>>(pooled, Wg, bg, probs_out, topi_out, wbe);

    // 8) init output with residual x1, then accumulate experts
    copy_f32_kernel<<<4096, 256, 0, stream>>>(x1, xout, X_ELEMS);

    dim3 gM1(NB * MT_PER_S, FF / 256);               // (416, 12)
    dim3 gM2(NB * MT_PER_S, DD / 256);               // (416, 3)
    for (int e = 0; e < NE; ++e) {
        gemm_moe1_kernel<<<gM1, 128, 0, stream>>>(
            nx2_bf, we1_bf + (size_t)e * DD * FF, be1 + (size_t)e * FF, h_bf, wbe, e);
        gemm_moe2_kernel<<<gM2, 128, 0, stream>>>(
            h_bf, we2_bf + (size_t)e * FF * DD, be2 + (size_t)e * DD, xout, wbe, e);
    }
}
